// MultiHeadSelfAttention_6382321402109
// MI455X (gfx1250) — matrix operand order repaired and checked by reading, not compiled, not run on hardware
//
#include <hip/hip_runtime.h>

#define D_MODEL 768
#define NHEADS  12
#define HD      64
#define SEQ     2048
#define BATCH   2
#define NTOK    (BATCH*SEQ)          // 4096
#define BHPAIRS (BATCH*NHEADS)       // 24

typedef __attribute__((ext_vector_type(16))) __bf16 v16bf;
typedef __attribute__((ext_vector_type(8)))  float  v8f;
typedef __attribute__((ext_vector_type(4)))  unsigned v4u;
typedef __attribute__((ext_vector_type(8)))  int    v8i_;
typedef __attribute__((ext_vector_type(4)))  int    v4i_;

union FragBF { v16bf v; unsigned u[8]; };

__device__ __forceinline__ unsigned short f2bf(float f) {
    unsigned u = __float_as_uint(f);
    u += 0x7FFFu + ((u >> 16) & 1u);          // round-to-nearest-even
    return (unsigned short)(u >> 16);
}

// TDM: stage a 32x32 bf16 tile (row stride D_MODEL) from global to LDS.
// D# per CDNA5 ISA ch.8: group0 = {count/type, lds_addr, global_addr},
// group1 = {data_size, tensor dims/strides, tile dims}, groups2/3 = 0 (2D).
// This toolchain exposes the 6-arg builtin (g0, g1, g2, g3, extra, cpol).
__device__ __forceinline__ void tdm_load_slab(const unsigned short* gsrc,
                                              unsigned ldsoff) {
    unsigned long long ga = (unsigned long long)(size_t)gsrc;
    v4u g0 = { 1u,                                   // count=1, user D#
               ldsoff,                               // lds_addr[31:0]
               (unsigned)ga,                         // global_addr[31:0]
               ((unsigned)(ga >> 32) & 0x01FFFFFFu)  // global_addr[56:32]
                   | (2u << 30) };                   // type=2 ("image")
    v8i_ g1 = { (int)(1u << 16),                     // data_size=1 (2 bytes)
                (int)(((unsigned)D_MODEL & 0xFFFFu) << 16),      // t_dim0 lo16
                (int)(((unsigned)NTOK & 0xFFFFu) << 16),         // t_dim0 hi | t_dim1 lo
                (int)(32u << 16),                    // t_dim1 hi | tile_dim0=32
                (int)32,                             // tile_dim1=32, tile_dim2=0
                (int)D_MODEL,                        // t_dim0_stride lo32
                0, 0 };                              // stride hi, dim1_stride
    v4i_ gz4 = { 0, 0, 0, 0 };
    v8i_ gz8 = { 0, 0, 0, 0, 0, 0, 0, 0 };
    __builtin_amdgcn_tensor_load_to_lds(g0, g1, gz4, gz4, gz8, 0);
}

// ---------------- conversion kernels ----------------
__global__ void k_f32_to_bf16(const float* __restrict__ src,
                              unsigned short* __restrict__ dst, int n) {
    int i = blockIdx.x * blockDim.x + threadIdx.x;
    if (i < n) dst[i] = f2bf(src[i]);
}

// dst[n][k] = bf16(src[k][n])   (768x768)
__global__ void k_transpose_to_bf16(const float* __restrict__ src,
                                    unsigned short* __restrict__ dst) {
    int i = blockIdx.x * blockDim.x + threadIdx.x;
    if (i < D_MODEL * D_MODEL) {
        int n = i / D_MODEL, k = i % D_MODEL;
        dst[i] = f2bf(src[(size_t)k * D_MODEL + n]);
    }
}

// ============================================================================
// GEMM: C[M=4096,N=768] = A(bf16,[M,K]) @ Wt(bf16,[N,K]) + bias
// Block = 128 threads = 4 waves. Block tile: 32(M) x 256(N).
// Wave tile: 32(M) x 64(N) = 8 accumulators -> 8 WMMA per 32-wide k-step.
// A k-slab (32x32 bf16, 2KB) moved by the Tensor Data Mover
// (tensor_load_to_lds, TENSORcnt), double buffered; wave 0 drives the TDM.
// mode 0: f32 row-major          (final projection)
// mode 1: bf16 -> [B,H,S,hd]     (Q, K)
// mode 2: bf16 -> [B,H,hd,S]     (V transposed)
// ============================================================================
__global__ void __launch_bounds__(128)
k_gemm(const unsigned short* __restrict__ A,
       const unsigned short* __restrict__ Wt,
       const float* __restrict__ bias,
       float* __restrict__ outF,
       unsigned short* __restrict__ outB,
       int mode)
{
    __shared__ unsigned short Albuf[2][32 * 32];   // double-buffered A k-slab

    const int lane  = threadIdx.x & 31;
    const int w     = threadIdx.x >> 5;            // wave in block: N sub-strip
    const int tileN = blockIdx.x % 3;              // 3 N-blocks of 256
    const int tileM = blockIdx.x / 3;              // 128 M-blocks of 32
    const int hf    = lane >> 4;
    const int l15   = lane & 15;

    const unsigned short* aslab = A + (size_t)(tileM * 32) * D_MODEL;

    // B row pointers: 4 N sub-tiles per wave
    const unsigned* Brow[4];
    int ncol[4];
#pragma unroll
    for (int t = 0; t < 4; ++t) {
        ncol[t] = tileN * 256 + w * 64 + 16 * t + l15;
        Brow[t] = (const unsigned*)(Wt + (size_t)ncol[t] * D_MODEL);
    }

    v8f acc[2][4];
    v8f zero = {};
#pragma unroll
    for (int mi = 0; mi < 2; ++mi)
#pragma unroll
        for (int t = 0; t < 4; ++t) acc[mi][t] = zero;

    // prologue: TDM stages slab 0 (wave 0 only; TDM is wave-scoped, EXEC-agnostic)
    if (w == 0)
        tdm_load_slab(aslab, (unsigned)(size_t)&Albuf[0][0]);

    const int NK = D_MODEL / 32;                   // 24 k-steps
    for (int ks = 0; ks < NK; ++ks) {
        if (w == 0) __builtin_amdgcn_s_wait_tensorcnt(0);
        __syncthreads();                           // slab visible block-wide

        if (w == 0 && ks + 1 < NK)                 // DMA next slab during math
            tdm_load_slab(aslab + (ks + 1) * 32,
                          (unsigned)(size_t)&Albuf[(ks + 1) & 1][0]);

        // A fragments from LDS (two 16-row fragments, ds_load_b128-aligned)
        const unsigned* albuf_u = (const unsigned*)&Albuf[ks & 1][0];
        FragBF a[2];
#pragma unroll
        for (int mi = 0; mi < 2; ++mi) {
            const int base = (mi * 16 + l15) * 16 + 8 * hf;   // dword index
#pragma unroll
            for (int v = 0; v < 4; ++v) {
                a[mi].u[v]     = albuf_u[base + v];
                a[mi].u[4 + v] = albuf_u[base + 4 + v];
            }
        }

        // B fragments from global (streamed, L2-resident weights)
        const int k0 = ks * 32;
        const int hb = (k0 >> 1) + 8 * hf;
#pragma unroll
        for (int t = 0; t < 4; ++t) {
            FragBF b;
            __builtin_prefetch(Brow[t] + hb + 16, 0, 0);
#pragma unroll
            for (int v = 0; v < 8; ++v) b.u[v] = Brow[t][hb + v];
            acc[0][t] = __builtin_amdgcn_wmma_f32_16x16x32_bf16(
                false, a[0].v, false, b.v, (short)0, acc[0][t], false, false);
            acc[1][t] = __builtin_amdgcn_wmma_f32_16x16x32_bf16(
                false, a[1].v, false, b.v, (short)0, acc[1][t], false, false);
        }
        __syncthreads();   // all waves done reading slab before next overwrite
    }

#pragma unroll
    for (int t = 0; t < 4; ++t) {
        const float bcol = bias[ncol[t]];
        const int hh = ncol[t] >> 6, d = ncol[t] & 63;
#pragma unroll
        for (int mi = 0; mi < 2; ++mi) {
#pragma unroll
            for (int r = 0; r < 8; ++r) {
                const int   m   = tileM * 32 + mi * 16 + 8 * hf + r;
                const float val = acc[mi][t][r] + bcol;
                if (mode == 0) {
                    outF[(size_t)m * D_MODEL + ncol[t]] = val;
                } else {
                    const int bb = m >> 11, s = m & (SEQ - 1);
                    size_t idx;
                    if (mode == 1) idx = (((size_t)(bb * NHEADS + hh)) * SEQ + s) * HD + d;
                    else           idx = (((size_t)(bb * NHEADS + hh)) * HD + d) * SEQ + s;
                    outB[idx] = f2bf(val);
                }
            }
        }
    }
}

// ---------------- flash attention: one wave per 16-query block ------------
__global__ void __launch_bounds__(128)
k_attn(const unsigned short* __restrict__ Q,   // [B,H,S,hd] bf16
       const unsigned short* __restrict__ Kb,  // [B,H,S,hd] bf16
       const unsigned short* __restrict__ Vt,  // [B,H,hd,S] bf16
       unsigned short* __restrict__ ctx)       // [B,S,D] bf16
{
    __shared__ unsigned short pst[4][16 * 32];   // per-wave P staging tile

    const int lane = threadIdx.x & 31;
    const int wib  = threadIdx.x >> 5;
    const int wave = blockIdx.x * 4 + wib;
    const int bh   = wave >> 7;                  // 128 q-blocks per (b,h)
    const int qblk = wave & 127;
    const int b    = bh / NHEADS, h = bh % NHEADS;
    const int hf   = lane >> 4, l15 = lane & 15;

    // Q fragments (16x64 -> two 16x32 A-fragments), loaded once
    const unsigned* qrow = (const unsigned*)(Q + ((size_t)bh * SEQ + qblk * 16 + l15) * HD);
    FragBF aq0, aq1;
    {
        const int qa = 8 * hf;
#pragma unroll
        for (int v = 0; v < 4; ++v) {
            aq0.u[v] = qrow[qa + v];      aq0.u[4 + v] = qrow[qa + 4 + v];
            aq1.u[v] = qrow[16 + qa + v]; aq1.u[4 + v] = qrow[16 + qa + 4 + v];
        }
    }

    float mrow[8], lrow[8];
    v8f acc[4];
    v8f zero = {};
#pragma unroll
    for (int r = 0; r < 8; ++r) { mrow[r] = -1e30f; lrow[r] = 0.f; }
#pragma unroll
    for (int t = 0; t < 4; ++t) acc[t] = zero;

    for (int kv0 = 0; kv0 < SEQ; kv0 += 32) {
        // K^T B-fragments: lane = kv column, halves = contiguous hd
        const unsigned* krow0 = (const unsigned*)(Kb + ((size_t)bh * SEQ + kv0 + l15) * HD);
        const unsigned* krow1 = krow0 + (16 * HD) / 2;   // +16 tokens
        FragBF bk00, bk01, bk10, bk11;
        const int kb = 8 * hf;
#pragma unroll
        for (int v = 0; v < 8; ++v) {
            bk00.u[v] = krow0[kb + v];  bk01.u[v] = krow0[16 + kb + v];
            bk10.u[v] = krow1[kb + v];  bk11.u[v] = krow1[16 + kb + v];
        }

        v8f s0 = __builtin_amdgcn_wmma_f32_16x16x32_bf16(false, aq0.v, false, bk00.v,
                                                         (short)0, zero, false, false);
        s0     = __builtin_amdgcn_wmma_f32_16x16x32_bf16(false, aq1.v, false, bk01.v,
                                                         (short)0, s0, false, false);
        v8f s1 = __builtin_amdgcn_wmma_f32_16x16x32_bf16(false, aq0.v, false, bk10.v,
                                                         (short)0, zero, false, false);
        s1     = __builtin_amdgcn_wmma_f32_16x16x32_bf16(false, aq1.v, false, bk11.v,
                                                         (short)0, s1, false, false);

        unsigned short* pb = &pst[wib][0];
#pragma unroll
        for (int r = 0; r < 8; ++r) {
            float a0 = s0[r] * 0.125f, a1 = s1[r] * 0.125f;   // 1/sqrt(64)
            float rm = fmaxf(a0, a1);
#pragma unroll
            for (int off = 1; off < 16; off <<= 1)
                rm = fmaxf(rm, __shfl_xor(rm, off, 32));
            const float mn    = fmaxf(mrow[r], rm);
            const float alpha = __expf(mrow[r] - mn);
            mrow[r] = mn;
            a0 = __expf(a0 - mn); a1 = __expf(a1 - mn);
            float rs = a0 + a1;
#pragma unroll
            for (int off = 1; off < 16; off <<= 1)
                rs += __shfl_xor(rs, off, 32);
            lrow[r] = lrow[r] * alpha + rs;
#pragma unroll
            for (int t = 0; t < 4; ++t) acc[t][r] *= alpha;
            const int m = 8 * hf + r;           // C-layout row
            pb[m * 32 + l15]      = f2bf(a0);   // k = lane&15
            pb[m * 32 + 16 + l15] = f2bf(a1);   // k = 16 + (lane&15)
        }

        // reload P as A-fragment (row-major [16][32] in LDS, same wave -> DScnt order)
        FragBF ap;
        const unsigned* prow = (const unsigned*)(pb + l15 * 32);
        const int pa = 8 * hf;
#pragma unroll
        for (int v = 0; v < 4; ++v) { ap.u[v] = prow[pa + v]; ap.u[4 + v] = prow[pa + 4 + v]; }

        // V B-fragments from transposed V: lane = hd column, halves contiguous in s
#pragma unroll
        for (int t = 0; t < 4; ++t) {
            FragBF bv;
            const unsigned* vrow =
                (const unsigned*)(Vt + ((size_t)bh * HD + 16 * t + l15) * SEQ);
            const int vi = (kv0 >> 1) + 8 * hf;
#pragma unroll
            for (int v = 0; v < 8; ++v) bv.u[v] = vrow[vi + v];
            acc[t] = __builtin_amdgcn_wmma_f32_16x16x32_bf16(false, ap.v, false, bv.v,
                                                             (short)0, acc[t], false, false);
        }
    }

#pragma unroll
    for (int r = 0; r < 8; ++r) {
        const float inv = 1.0f / lrow[r];
        const int m = 8 * hf + r;
        const size_t base = ((size_t)(b * SEQ + qblk * 16 + m)) * D_MODEL + h * HD;
#pragma unroll
        for (int t = 0; t < 4; ++t)
            ctx[base + 16 * t + l15] = f2bf(acc[t][r] * inv);
    }
}

// ---------------- host launcher ----------------
extern "C" void kernel_launch(void* const* d_in, const int* in_sizes, int n_in,
                              void* d_out, int out_size, void* d_ws, size_t ws_size,
                              hipStream_t stream) {
    const float* x  = (const float*)d_in[0];
    const float* Wq = (const float*)d_in[1];
    const float* bq = (const float*)d_in[2];
    const float* Wk = (const float*)d_in[3];
    const float* bk = (const float*)d_in[4];
    const float* Wv = (const float*)d_in[5];
    const float* bv = (const float*)d_in[6];
    const float* Wo = (const float*)d_in[7];
    const float* bo = (const float*)d_in[8];
    float* out = (float*)d_out;

    unsigned short* xb   = (unsigned short*)d_ws;
    unsigned short* wqt  = xb  + (size_t)NTOK * D_MODEL;
    unsigned short* wkt  = wqt + (size_t)D_MODEL * D_MODEL;
    unsigned short* wvt  = wkt + (size_t)D_MODEL * D_MODEL;
    unsigned short* wot  = wvt + (size_t)D_MODEL * D_MODEL;
    unsigned short* Qb   = wot + (size_t)D_MODEL * D_MODEL;
    unsigned short* Kbf  = Qb  + (size_t)NTOK * D_MODEL;
    unsigned short* Vtb  = Kbf + (size_t)NTOK * D_MODEL;
    unsigned short* ctxb = Vtb + (size_t)NTOK * D_MODEL;

    const int nx = NTOK * D_MODEL;          // 3,145,728
    k_f32_to_bf16<<<nx / 256, 256, 0, stream>>>(x, xb, nx);
    const int nw = D_MODEL * D_MODEL;       // 589,824
    k_transpose_to_bf16<<<nw / 256, 256, 0, stream>>>(Wq, wqt);
    k_transpose_to_bf16<<<nw / 256, 256, 0, stream>>>(Wk, wkt);
    k_transpose_to_bf16<<<nw / 256, 256, 0, stream>>>(Wv, wvt);
    k_transpose_to_bf16<<<nw / 256, 256, 0, stream>>>(Wo, wot);

    const int gemm_blocks = (NTOK / 32) * (D_MODEL / 256);   // 128*3 = 384
    k_gemm<<<gemm_blocks, 128, 0, stream>>>(xb, wqt, bq, nullptr, Qb,  1);
    k_gemm<<<gemm_blocks, 128, 0, stream>>>(xb, wkt, bk, nullptr, Kbf, 1);
    k_gemm<<<gemm_blocks, 128, 0, stream>>>(xb, wvt, bv, nullptr, Vtb, 2);

    const int attn_blocks = (BHPAIRS * (SEQ / 16)) / 4;   // 768 blocks
    k_attn<<<attn_blocks, 128, 0, stream>>>(Qb, Kbf, Vtb, ctxb);

    k_gemm<<<gemm_blocks, 128, 0, stream>>>(ctxb, wot, bo, out, nullptr, 0);
}
